// NonLocalBlock2D_23244363006290
// MI455X (gfx1250) — matrix operand order repaired and checked by reading, not compiled, not run on hardware
//
#include <hip/hip_runtime.h>
#include <hip/hip_bf16.h>
#include <math.h>

// ---------------------------------------------------------------------------
// NonLocalBlock2D for MI455X (gfx1250): flash-attention with v_wmma_f32_16x16x32_f16
// B=8, C=256, CI=128, N=H*W=4096
// ---------------------------------------------------------------------------

typedef __attribute__((ext_vector_type(16))) _Float16 v16h;
typedef __attribute__((ext_vector_type(8)))  _Float16 v8h;
typedef __attribute__((ext_vector_type(8)))  float    v8f;

#define NB  8
#define NC  256
#define NCI 128
#define NN  4096

union HF16 { v16h v; v8h h2[2]; };

// A-fragment (16x32 f16, MxK): lane L holds row m=L%16; with h8=L/16 the 16
// halfs cover K = {k0+h8*8 .. +7} and {k0+16+h8*8 .. +7} (ISA 7.12.2 layout).
__device__ __forceinline__ v16h load_a_frag(const _Float16* base, int stride_h,
                                            int row, int k0, int h8) {
  const _Float16* p = base + (size_t)row * stride_h + k0 + h8 * 8;
  HF16 u;
  u.h2[0] = *(const v8h*)(p);
  u.h2[1] = *(const v8h*)(p + 16);
  return u.v;
}

// B-fragment (32x16 f16, KxN): lane L holds column n=L%16; halfs cover
// K = {k0+h8*8 .. +7} and {k0+16+h8*8 .. +7} (same as A-fragment).
__device__ __forceinline__ v16h load_b_frag(const _Float16* base, int stride_h,
                                            int colrow, int k0, int h8) {
  const _Float16* p = base + (size_t)colrow * stride_h + k0 + h8 * 8;
  HF16 u;
  u.h2[0] = *(const v8h*)(p);
  u.h2[1] = *(const v8h*)(p + 16);
  return u.v;
}

__device__ __forceinline__ v8f wmma16(v16h a, v16h b, v8f c) {
  return __builtin_amdgcn_wmma_f32_16x16x32_f16(false, a, false, b,
                                                (short)0, c, false, false);
}

// ---------------------------------------------------------------------------
// Kernel 0: cast all weights f32 -> f16 into workspace (each 128*256 elems).
// ---------------------------------------------------------------------------
__global__ void cvt_weights(const float* __restrict__ Wt,
                            const float* __restrict__ Wp,
                            const float* __restrict__ Wg,
                            const float* __restrict__ Wm,
                            _Float16* __restrict__ dst) {
  int i = blockIdx.x * blockDim.x + threadIdx.x;
  if (i < NCI * NC) {
    dst[i]                = (_Float16)Wt[i];
    dst[NCI * NC + i]     = (_Float16)Wp[i];
    dst[2 * NCI * NC + i] = (_Float16)Wg[i];
    dst[3 * NCI * NC + i] = (_Float16)Wm[i];
  }
}

// ---------------------------------------------------------------------------
// Kernel 1: input projections. Per block: one batch b, 64 positions.
// theta/phi -> [B][N][CI] f16 (row-major over CI); g -> [B][CI][N] f16.
// p-loop fully unrolled so weight/bias/dst pointers are compile-time selected
// (no v_cndmask pointer blends in the inner loop).
// ---------------------------------------------------------------------------
__global__ __launch_bounds__(256) void proj_kernel(
    const float* __restrict__ x,
    const _Float16* __restrict__ Wt16, const _Float16* __restrict__ Wp16,
    const _Float16* __restrict__ Wg16,
    const float* __restrict__ bt, const float* __restrict__ bp,
    const float* __restrict__ bg,
    _Float16* __restrict__ theta16, _Float16* __restrict__ phi16,
    _Float16* __restrict__ g16) {
  __shared__ _Float16 xT[64][NC + 8];   // stride 264 halfs = 528B (16B multiple)
  const int b  = blockIdx.y;
  const int n0 = blockIdx.x * 64;
  const int tid = threadIdx.x;

  // stage x tile (f32 -> f16, transposed to [n_local][c])
  for (int idx = tid; idx < 64 * NC; idx += 256) {
    int c = idx >> 6;
    int i = idx & 63;
    xT[i][c] = (_Float16)x[((size_t)b * NC + c) * NN + n0 + i];
  }
  __syncthreads();

  const int wave = tid >> 5;
  const int lane = tid & 31;
  const int l16  = lane & 15;
  const int h8   = lane >> 4;

#pragma unroll
  for (int p = 0; p < 3; ++p) {
    const _Float16* W16 = (p == 0) ? Wt16 : (p == 1) ? Wp16 : Wg16;
    const float* bias   = (p == 0) ? bt   : (p == 1) ? bp   : bg;
    // 32 tiles per projection (4 n-rows x 8 CI-cols), 4 per wave
    for (int i = 0; i < 4; ++i) {
      int tile = wave + i * 8;
      int rt = tile >> 3;
      int ct = tile & 7;
      v8f acc = {};
#pragma unroll
      for (int k = 0; k < NC; k += 32) {
        v16h a  = load_a_frag(&xT[rt * 16][0], NC + 8, l16, k, h8);
        v16h bf = load_b_frag(W16, NC, ct * 16 + l16, k, h8);
        acc = wmma16(a, bf, acc);
      }
      float bv = bias[ct * 16 + l16];
      if (p < 2) {
        _Float16* dst = ((p == 0) ? theta16 : phi16) +
                        ((size_t)b * NN + n0 + rt * 16) * NCI + ct * 16 + l16;
#pragma unroll
        for (int r = 0; r < 8; ++r)
          dst[(size_t)(r + h8 * 8) * NCI] = (_Float16)(acc[r] + bv);
      } else {
        _Float16* dst = g16 + ((size_t)b * NCI + ct * 16 + l16) * NN +
                        n0 + rt * 16 + h8 * 8;
#pragma unroll
        for (int r = 0; r < 8; ++r)
          dst[r] = (_Float16)(acc[r] + bv);
      }
    }
  }
}

// ---------------------------------------------------------------------------
// Kernel 2: flash attention, 64-key KV tiles. 4 waves/block, each wave owns 16
// query rows. S = Q K^T (WMMA), online max via shfl_xor inside 16-lane halves,
// row-sums computed by an extra WMMA against a ones-fragment (C-tile layout is
// exactly the per-lane softmax-state layout), P transposed C->A through
// per-wave LDS, Y += P V (WMMA). y -> [B][N][CI] f16.
// ---------------------------------------------------------------------------
__global__ __launch_bounds__(128) void attn_kernel(
    const _Float16* __restrict__ theta16, const _Float16* __restrict__ phi16,
    const _Float16* __restrict__ g16, _Float16* __restrict__ y16) {
  __shared__ _Float16 Pld[4][16][72];   // per-wave 16x64 P tile, stride 72 halfs
  const int b    = blockIdx.y;
  const int wave = threadIdx.x >> 5;
  const int lane = threadIdx.x & 31;
  const int l16  = lane & 15;
  const int h8   = lane >> 4;
  const int qn0  = blockIdx.x * 64 + wave * 16;

  const _Float16* thB = theta16 + (size_t)b * NN * NCI;
  const _Float16* phB = phi16   + (size_t)b * NN * NCI;
  const _Float16* gB  = g16     + (size_t)b * NCI * NN;

  v16h Qa[4];
#pragma unroll
  for (int k = 0; k < 4; ++k)
    Qa[k] = load_a_frag(thB + (size_t)qn0 * NCI, NCI, l16, k * 32, h8);

  v16h onesb;
#pragma unroll
  for (int i = 0; i < 16; ++i) onesb[i] = (_Float16)1.0f;

  v8f Y[8] = {};
  v8f Yl = {};            // row-sum accumulator (every column identical)
  float rm[8];
#pragma unroll
  for (int r = 0; r < 8; ++r) rm[r] = -1.0e30f;

  _Float16 (*P)[72] = Pld[wave];

#pragma unroll 1
  for (int j0 = 0; j0 < NN; j0 += 64) {
    if (j0 + 64 < NN) {
      __builtin_prefetch(phB + (size_t)(j0 + 64 + lane) * NCI, 0, 1);
      __builtin_prefetch(phB + (size_t)(j0 + 96 + lane) * NCI, 0, 1);
      __builtin_prefetch(gB + (size_t)(lane * 4) * NN + j0 + 64, 0, 1);
    }
    // 4 S tiles covering keys [j0, j0+64)
    v8f s[4] = {};
#pragma unroll
    for (int st = 0; st < 4; ++st)
#pragma unroll
      for (int k = 0; k < 4; ++k)
        s[st] = wmma16(Qa[k],
                       load_b_frag(phB, NCI, j0 + st * 16 + l16, k * 32, h8),
                       s[st]);

    // running row max (rows live across the 16 lanes of each half)
    float tmax[8];
#pragma unroll
    for (int r = 0; r < 8; ++r)
      tmax[r] = fmaxf(fmaxf(s[0][r], s[1][r]), fmaxf(s[2][r], s[3][r]));
#pragma unroll
    for (int m = 8; m >= 1; m >>= 1)
#pragma unroll
      for (int r = 0; r < 8; ++r)
        tmax[r] = fmaxf(tmax[r], __shfl_xor(tmax[r], m, 32));

    float sc[8];
#pragma unroll
    for (int r = 0; r < 8; ++r) {
      float nm = fmaxf(rm[r], tmax[r]);
      sc[r] = __expf(rm[r] - nm);
      rm[r] = nm;
    }
    // rescale accumulators (incl. row-sum tile)
#pragma unroll
    for (int t = 0; t < 8; ++t)
#pragma unroll
      for (int r = 0; r < 8; ++r) Y[t][r] *= sc[r];
#pragma unroll
    for (int r = 0; r < 8; ++r) Yl[r] *= sc[r];

    // P = exp(S - m), written to per-wave LDS (C layout -> A layout transpose)
#pragma unroll
    for (int st = 0; st < 4; ++st)
#pragma unroll
      for (int r = 0; r < 8; ++r)
        P[r + h8 * 8][st * 16 + l16] = (_Float16)__expf(s[st][r] - rm[r]);
    asm volatile("s_wait_dscnt 0" ::: "memory");   // same-wave DS RAW
    v16h pa0 = load_a_frag(&P[0][0], 72, l16, 0, h8);
    v16h pa1 = load_a_frag(&P[0][0], 72, l16, 32, h8);

    // row sums via WMMA against ones; Y += P * V (V = g[b][d][j])
    Yl = wmma16(pa0, onesb, Yl);
    Yl = wmma16(pa1, onesb, Yl);
#pragma unroll
    for (int t = 0; t < 8; ++t)
      Y[t] = wmma16(pa0, load_b_frag(gB, NN, t * 16 + l16, j0, h8), Y[t]);
#pragma unroll
    for (int t = 0; t < 8; ++t)
      Y[t] = wmma16(pa1, load_b_frag(gB, NN, t * 16 + l16, j0 + 32, h8), Y[t]);
  }

  float inv[8];
#pragma unroll
  for (int r = 0; r < 8; ++r) inv[r] = 1.0f / Yl[r];
#pragma unroll
  for (int t = 0; t < 8; ++t) {
    _Float16* dst = y16 + ((size_t)b * NN + qn0 + h8 * 8) * NCI + t * 16 + l16;
#pragma unroll
    for (int r = 0; r < 8; ++r)
      dst[(size_t)r * NCI] = (_Float16)(Y[t][r] * inv[r]);
  }
}

// ---------------------------------------------------------------------------
// Kernel 3: output projection + bias + residual. out[b][c][n] (f32).
// ---------------------------------------------------------------------------
__global__ __launch_bounds__(256) void out_kernel(
    const _Float16* __restrict__ y16, const _Float16* __restrict__ Wm16,
    const float* __restrict__ bm, const float* __restrict__ x,
    float* __restrict__ out) {
  const int b    = blockIdx.y;
  const int n0   = blockIdx.x * 64;
  const int wave = threadIdx.x >> 5;
  const int lane = threadIdx.x & 31;
  const int l16  = lane & 15;
  const int h8   = lane >> 4;
  const _Float16* yB = y16 + (size_t)b * NN * NCI;

  for (int it = 0; it < 8; ++it) {
    int t  = it * 8 + wave;    // 64 tiles: 4 n-rows x 16 c-cols
    int rt = t >> 4;
    int ct = t & 15;
    v8f acc = {};
#pragma unroll
    for (int k = 0; k < NCI; k += 32) {
      v16h a  = load_a_frag(yB + (size_t)(n0 + rt * 16) * NCI, NCI, l16, k, h8);
      v16h bf = load_b_frag(Wm16, NCI, ct * 16 + l16, k, h8);
      acc = wmma16(a, bf, acc);
    }
    int c = ct * 16 + l16;
    float bv = bm[c];
    size_t base = ((size_t)b * NC + c) * NN + n0 + rt * 16 + h8 * 8;
#pragma unroll
    for (int r = 0; r < 8; ++r)
      out[base + r] = acc[r] + bv + x[base + r];
  }
}

// ---------------------------------------------------------------------------
extern "C" void kernel_launch(void* const* d_in, const int* in_sizes, int n_in,
                              void* d_out, int out_size, void* d_ws, size_t ws_size,
                              hipStream_t stream) {
  (void)in_sizes; (void)n_in; (void)out_size; (void)ws_size;
  const float* x  = (const float*)d_in[0];
  const float* Wg = (const float*)d_in[1];
  const float* bg = (const float*)d_in[2];
  const float* Wt = (const float*)d_in[3];
  const float* bt = (const float*)d_in[4];
  const float* Wp = (const float*)d_in[5];
  const float* bp = (const float*)d_in[6];
  const float* Wm = (const float*)d_in[7];
  const float* bm = (const float*)d_in[8];

  _Float16* Wt16 = (_Float16*)d_ws;
  _Float16* Wp16 = Wt16 + 32768;
  _Float16* Wg16 = Wp16 + 32768;
  _Float16* Wm16 = Wg16 + 32768;
  _Float16* th16 = Wm16 + 32768;                   // [B][N][CI]
  _Float16* ph16 = th16 + (size_t)NB * NN * NCI;   // [B][N][CI]
  _Float16* g16  = ph16 + (size_t)NB * NN * NCI;   // [B][CI][N]
  _Float16* y16  = g16  + (size_t)NB * NN * NCI;   // [B][N][CI]

  cvt_weights<<<dim3(128), dim3(256), 0, stream>>>(Wt, Wp, Wg, Wm, Wt16);
  proj_kernel<<<dim3(NN / 64, NB), dim3(256), 0, stream>>>(
      x, Wt16, Wp16, Wg16, bt, bp, bg, th16, ph16, g16);
  attn_kernel<<<dim3(NN / 64, NB), dim3(128), 0, stream>>>(th16, ph16, g16, y16);
  out_kernel<<<dim3(NN / 64, NB), dim3(256), 0, stream>>>(y16, Wm16, bm, x,
                                                          (float*)d_out);
}
